// SingleHeadDilatedSelfAttention_18726057411206
// MI455X (gfx1250) — matrix operand order repaired and checked by reading, not compiled, not run on hardware
//
#include <hip/hip_runtime.h>
#include <math.h>
#include <stdint.h>

typedef _Float16 half8  __attribute__((ext_vector_type(8)));
typedef _Float16 half16 __attribute__((ext_vector_type(16)));
typedef float    float8 __attribute__((ext_vector_type(8)));

#define B_    8
#define T_    16384
#define E_    512
#define SEG_  16
#define LD_   512          // dilated rows per segment (L / DIL)
#define DK_   64
#define DV_   64
#define NORM_ 0.125f       // 1/sqrt(64)
#define POSL_ 1024         // positional embedding leading dim (L)

// f16 workspace: Q | K | V, each B*SEG*LD_*DK_ halves (8 MB each)
#define QKOFF ((size_t)B_ * SEG_ * LD_ * DK_)

#define SMEM2 (288 * 1024)

__device__ __forceinline__ half16 cat8(half8 lo, half8 hi) {
  return __builtin_shufflevector(lo, hi, 0,1,2,3,4,5,6,7,8,9,10,11,12,13,14,15);
}

__device__ __forceinline__ float8 f8zero() {
  float8 z = {0.f, 0.f, 0.f, 0.f, 0.f, 0.f, 0.f, 0.f};
  return z;
}

// Async global->LDS 16-byte copy (CDNA5 path, ASYNCcnt-tracked): bypasses the
// VGPR round trip of load_b128 + ds_store_b128 and frees VMEM/VALU slots.
__device__ __forceinline__ void async_copy16(unsigned lds_off, const void* gaddr) {
  asm volatile("global_load_async_to_lds_b128 %0, %1, off"
               :: "v"(lds_off), "v"(gaddr) : "memory");
}
__device__ __forceinline__ void wait_asynccnt0() {
  asm volatile("s_wait_asynccnt 0x0" ::: "memory");
}

// ---------------------------------------------------------------------------
// Kernel 1: fused Q/K/V projection.
// grid.x = B*SEG*4; each block: one (b,s), 128 of the 512 dilated rows.
// 8 waves; wave w owns rows [16w, 16w+16); output tile is 16 x 192
// (cols 0..63 -> Q, 64..127 -> K, 128..191 -> V), K-loop over E in 32-chunks.
// ---------------------------------------------------------------------------
__global__ __launch_bounds__(256) void qkv_proj_wmma(
    const float* __restrict__ x,
    const float* __restrict__ Wq,
    const float* __restrict__ Wk,
    const float* __restrict__ Wv,
    _Float16* __restrict__ qkv)
{
  __shared__ __align__(64) _Float16 xs[128][32];
  __shared__ __align__(64) _Float16 wsm[192][32];

  const int bs = blockIdx.x >> 2;     // b*16 + s
  const int rb = blockIdx.x & 3;      // row block (128 rows each)
  const int b  = bs >> 4;
  const int s  = bs & 15;
  const int tid  = threadIdx.x;
  const int wave = tid >> 5;
  const int lane = tid & 31;
  const int r    = lane & 15;
  const int hi   = lane >> 4;

  float8 acc[12];
#pragma unroll
  for (int i = 0; i < 12; ++i) acc[i] = f8zero();

  for (int ec = 0; ec < 16; ++ec) {
    // Stage dilated X rows: row m -> x[b, 32*m + s, ec*32 .. ec*32+31], fp32->f16
#pragma unroll
    for (int u = tid; u < 128 * 8; u += 256) {
      const int row = u >> 3;
      const int c4  = (u & 7) * 4;
      const int t   = 32 * (128 * rb + row) + s;
      const float4 f = *(const float4*)&x[((size_t)b * T_ + t) * E_ + ec * 32 + c4];
      xs[row][c4 + 0] = (_Float16)f.x;
      xs[row][c4 + 1] = (_Float16)f.y;
      xs[row][c4 + 2] = (_Float16)f.z;
      xs[row][c4 + 3] = (_Float16)f.w;
    }
    // Stage fused weights [Wq; Wk; Wv] (192 x 32 chunk), fp32->f16
#pragma unroll
    for (int u = tid; u < 192 * 8; u += 256) {
      const int row = u >> 3;
      const int c4  = (u & 7) * 4;
      const float* src = (row < 64)  ? &Wq[(size_t)row * E_]
                       : (row < 128) ? &Wk[(size_t)(row - 64) * E_]
                                     : &Wv[(size_t)(row - 128) * E_];
      const float4 f = *(const float4*)&src[ec * 32 + c4];
      wsm[row][c4 + 0] = (_Float16)f.x;
      wsm[row][c4 + 1] = (_Float16)f.y;
      wsm[row][c4 + 2] = (_Float16)f.z;
      wsm[row][c4 + 3] = (_Float16)f.w;
    }
    __syncthreads();

    // A fragment (16x32 f16): lane r holds row M=r; K = hi*16 + j (contiguous)
    const int rowl = wave * 16 + r;
    half8 a0 = *(const half8*)&xs[rowl][hi * 16];
    half8 a1 = *(const half8*)&xs[rowl][hi * 16 + 8];
    half16 a = cat8(a0, a1);

#pragma unroll
    for (int nt = 0; nt < 12; ++nt) {
      // B fragment (32x16): lane -> N = r (output channel nt*16+r), halves -> K = hi*16+j
      half16 bf = *(const half16*)&wsm[nt * 16 + r][hi * 16];
      acc[nt] = __builtin_amdgcn_wmma_f32_16x16x32_f16(
          false, a, false, bf, (short)0, acc[nt], false, false);
    }
    __syncthreads();
  }

  // Store Q/K/V as f16 to workspace.  D layout: VGPR i -> M = i + 8*hi, lane -> N = r.
  const int mbase = 128 * rb + wave * 16;
#pragma unroll
  for (int nt = 0; nt < 12; ++nt) {
    const int col = nt * 16 + r;
#pragma unroll
    for (int i = 0; i < 8; ++i) {
      const int m = mbase + hi * 8 + i;
      const float v = acc[nt][i];
      const size_t idx = ((size_t)bs * LD_ + m) * DK_;
      if (col < 64)        qkv[idx + col]                    = (_Float16)v;
      else if (col < 128)  qkv[QKOFF + idx + (col - 64)]     = (_Float16)v;
      else                 qkv[2 * QKOFF + idx + (col - 128)] = (_Float16)v;
    }
  }
}

// ---------------------------------------------------------------------------
// Kernel 2: dilated attention per (b,s); grid.x = B*SEG, 8 waves.
// LDS (288 KB dynamic): Kh[512][64]f16 | Qh[512][64]f16 | Vt[64][512]f16
//                       | Sb[32][512]f32 | St[512][32]f16 (smS transposed)
// ---------------------------------------------------------------------------
__global__ __launch_bounds__(256) void dil_attn_wmma(
    const _Float16* __restrict__ qkv,
    const float* __restrict__ pos,
    float* __restrict__ out)
{
  extern __shared__ char smem[];
  _Float16* Kh = (_Float16*)smem;            // 64 KB
  _Float16* Qh = Kh + 512 * 64;              // 64 KB
  _Float16* Vt = Qh + 512 * 64;              // 64 KB (transposed V: [d][n])
  float*    Sb = (float*)(Vt + 64 * 512);    // 64 KB (32 x 512 score rows)
  _Float16* St = (_Float16*)(Sb + 32 * 512); // 32 KB (smS^T: [m][n-in-block])

  const int bs = blockIdx.x;
  const int b  = bs >> 4;
  const int s  = bs & 15;
  const int tid  = threadIdx.x;
  const int wave = tid >> 5;
  const int lane = tid & 31;
  const int r    = lane & 15;
  const int hi   = lane >> 4;

  const _Float16* qg = qkv + ((size_t)bs * LD_) * DK_;
  const _Float16* kg = qkv + QKOFF + ((size_t)bs * LD_) * DK_;
  const _Float16* vg = qkv + 2 * QKOFF + ((size_t)bs * LD_) * DK_;

  // LDS byte offsets for async global->LDS copies (low 32 bits of generic ptr)
  const unsigned KhOff = (unsigned)(uintptr_t)Kh;
  const unsigned QhOff = (unsigned)(uintptr_t)Qh;

  // Phase A: K and Q are layout-identical f16 copies -> async DMA into LDS.
  // V needs a transpose -> regular load + LDS scatter.
#pragma unroll
  for (int c = tid; c < 512 * 8; c += 256) {
    const int m  = c >> 3;
    const int d0 = (c & 7) * 8;
    const unsigned boff = (unsigned)(m * 64 + d0) * 2u;   // bytes
    async_copy16(KhOff + boff, &kg[(size_t)m * 64 + d0]);
    async_copy16(QhOff + boff, &qg[(size_t)m * 64 + d0]);
    half8 v = *(const half8*)&vg[(size_t)m * 64 + d0];
#pragma unroll
    for (int j = 0; j < 8; ++j) Vt[(d0 + j) * 512 + m] = v[j];
  }
  wait_asynccnt0();
  __syncthreads();

  // att accumulators: wave w owns m in [64w, 64w+64) x d in [0,64): 16 tiles.
  float8 aacc[16];
#pragma unroll
  for (int i = 0; i < 16; ++i) aacc[i] = f8zero();

  for (int nb = 0; nb < 16; ++nb) {
    const int n0 = nb * 32;

    // Prefetch next block's positional-bias rows into cache (streamed f32).
    if (nb < 15) {
      const int ngn = 2 * (n0 + 32 + lane);
      __builtin_prefetch(&pos[(size_t)ngn * POSL_], 0, 1);
    }

    // --- S block (32 K-rows x 512 m-cols) = K . Q^T ---
    // wave w: rows (w&1)*16..+16, cols (w>>1)*128..+128 (8 tiles)
    const int r0l = (wave & 1) * 16;
    const int c0  = (wave >> 1) * 128;
    float8 sacc[8];
#pragma unroll
    for (int i = 0; i < 8; ++i) sacc[i] = f8zero();

#pragma unroll
    for (int dc = 0; dc < 64; dc += 32) {
      half8 a0 = *(const half8*)&Kh[(n0 + r0l + r) * 64 + dc + hi * 16];
      half8 a1 = *(const half8*)&Kh[(n0 + r0l + r) * 64 + dc + hi * 16 + 8];
      half16 a = cat8(a0, a1);
#pragma unroll
      for (int nt = 0; nt < 8; ++nt) {
        const int m0 = c0 + nt * 16;
        // B[K=dd][N=m'] = Q[m0+m'][dc+dd] -> contiguous 16 halves from Qh row
        half16 bf = *(const half16*)&Qh[(m0 + r) * 64 + dc + hi * 16];
        sacc[nt] = __builtin_amdgcn_wmma_f32_16x16x32_f16(
            false, a, false, bf, (short)0, sacc[nt], false, false);
      }
    }

    // mask (keep k >= m), add dilated positional bias, scale; spill rows to LDS
#pragma unroll
    for (int nt = 0; nt < 8; ++nt) {
      const int m = c0 + nt * 16 + r;
#pragma unroll
      for (int i = 0; i < 8; ++i) {
        const int nl = r0l + hi * 8 + i;
        const int ng = n0 + nl;
        float v = -__builtin_inff();
        if (ng >= m)
          v = (sacc[nt][i] + pos[(size_t)(2 * ng) * POSL_ + 2 * m]) * NORM_;
        Sb[nl * 512 + m] = v;
      }
    }
    __syncthreads();

    // --- softmax over m (axis=3, faithful): wave w handles rows 4w..4w+3 ---
#pragma unroll
    for (int rr = 0; rr < 4; ++rr) {
      const int nl = wave * 4 + rr;
      float vals[16];
      float mx = -__builtin_inff();
#pragma unroll
      for (int k = 0; k < 16; ++k) {
        vals[k] = Sb[nl * 512 + lane + 32 * k];
        mx = fmaxf(mx, vals[k]);
      }
#pragma unroll
      for (int off = 16; off > 0; off >>= 1)
        mx = fmaxf(mx, __shfl_xor(mx, off, 32));
      float sum = 0.f;
#pragma unroll
      for (int k = 0; k < 16; ++k) { vals[k] = __expf(vals[k] - mx); sum += vals[k]; }
#pragma unroll
      for (int off = 16; off > 0; off >>= 1)
        sum += __shfl_xor(sum, off, 32);
      const float inv = 1.f / sum;
#pragma unroll
      for (int k = 0; k < 16; ++k)
        St[(lane + 32 * k) * 32 + nl] = (_Float16)(vals[k] * inv);
    }
    __syncthreads();

    // --- att += smS^T . V over this 32-row block ---
#pragma unroll
    for (int mt = 0; mt < 4; ++mt) {
      const int mrow = wave * 64 + mt * 16 + r;
      half8 a0 = *(const half8*)&St[mrow * 32 + hi * 16];
      half8 a1 = *(const half8*)&St[mrow * 32 + hi * 16 + 8];
      half16 a = cat8(a0, a1);
#pragma unroll
      for (int dt = 0; dt < 4; ++dt) {
        // B[K=n'][N=d'] = V[n0+n'][d'] = Vt[d'][n0+n'] -> contiguous halves
        half16 bf = *(const half16*)&Vt[(dt * 16 + r) * 512 + n0 + hi * 16];
        aacc[mt * 4 + dt] = __builtin_amdgcn_wmma_f32_16x16x32_f16(
            false, a, false, bf, (short)0, aacc[mt * 4 + dt], false, false);
      }
    }
    __syncthreads();
  }

  // Scatter: out[b, 2*(s*512+m), d] = att[m][d]; odd time rows = 0.
#pragma unroll
  for (int mt = 0; mt < 4; ++mt) {
#pragma unroll
    for (int dt = 0; dt < 4; ++dt) {
      const int d = dt * 16 + r;
#pragma unroll
      for (int i = 0; i < 8; ++i) {
        const int m = wave * 64 + mt * 16 + hi * 8 + i;
        const int t = 2 * (s * LD_ + m);
        const size_t o = ((size_t)b * T_ + t) * DV_ + d;
        out[o]       = aacc[mt * 4 + dt][i];
        out[o + DV_] = 0.f;   // odd time index, zero-fill ('same' padding)
      }
    }
  }
}

// ---------------------------------------------------------------------------
extern "C" void kernel_launch(void* const* d_in, const int* in_sizes, int n_in,
                              void* d_out, int out_size, void* d_ws, size_t ws_size,
                              hipStream_t stream) {
  (void)in_sizes; (void)n_in; (void)out_size; (void)ws_size;
  const float* x   = (const float*)d_in[0];   // x_in (B,T,E)
  const float* Wk  = (const float*)d_in[1];   // (DK,E)
  const float* Wq  = (const float*)d_in[2];   // (DK,E)
  const float* Wv  = (const float*)d_in[3];   // (DV,E)
  const float* pos = (const float*)d_in[4];   // (L,L)
  float* out = (float*)d_out;
  _Float16* qkv = (_Float16*)d_ws;            // 3 * 8 MB f16 workspace

  (void)hipFuncSetAttribute(reinterpret_cast<const void*>(dil_attn_wmma),
                            hipFuncAttributeMaxDynamicSharedMemorySize, SMEM2);

  qkv_proj_wmma<<<B_ * SEG_ * 4, 256, 0, stream>>>(x, Wq, Wk, Wv, qkv);
  dil_attn_wmma<<<B_ * SEG_, 256, SMEM2, stream>>>(qkv, pos, out);
}
